// EmbedderNeuronGroup_index_45981919871512
// MI455X (gfx1250) — matrix operand order repaired and checked by reading, not compiled, not run on hardware
//
#include <hip/hip_runtime.h>
#include <hip/hip_bf16.h>

// CDNA5 / gfx1250: wave32, v_wmma_f32_16x16x32_bf16 (f32 accumulate).
// Fused gather + GEMM: per layer C[(b,k),o] = [x[b, start+k*w : +w] ++ x[b, start+kn*w+k]] @ W + bias.

typedef __attribute__((ext_vector_type(16))) __bf16 v16bf;
typedef __attribute__((ext_vector_type(8)))  __bf16 v8bf;
typedef __attribute__((ext_vector_type(8)))  float  v8f;

#define TOTAL_IN  97440
#define NGROUPS   240
#define DMODEL    1024
#define LDS_K     40      // 32 K-elements + 8 pad (bf16); 80B row stride (16B-aligned)

// ---- staging loads into registers (branch-free tail via address clamp) ----
__device__ __forceinline__ void load_a16(const float* __restrict__ x,
                                         long abase, long abias, int w,
                                         int kbase, bool tail, float va[16])
{
    if (!tail) {
#pragma unroll
        for (int i = 0; i < 16; ++i) va[i] = x[abase + kbase + i];
    } else {
#pragma unroll
        for (int i = 0; i < 16; ++i) {
            const int  col  = kbase + i;
            const long addr = (col < w) ? (abase + col) : abias;   // v_cndmask on address
            const float v   = x[addr];                              // unconditional load
            va[i] = (col <= w) ? v : 0.f;                           // zero K-padding
        }
    }
}

__device__ __forceinline__ void load_b16(const float* __restrict__ W,
                                         int i_dim, int bcol,
                                         int kbase, bool tail, float vb[16])
{
    if (!tail) {
#pragma unroll
        for (int i = 0; i < 16; ++i) vb[i] = W[(long)(kbase + i) * DMODEL + bcol];
    } else {
#pragma unroll
        for (int i = 0; i < 16; ++i) {
            const int gk  = kbase + i;
            const int gkc = (gk < i_dim) ? gk : (i_dim - 1);        // clamp row
            const float v = W[(long)gkc * DMODEL + bcol];
            vb[i] = (gk < i_dim) ? v : 0.f;
        }
    }
}

__global__ __launch_bounds__(256)
void fused_gather_gemm_bf16(const float* __restrict__ x,
                            const float* __restrict__ W,
                            const float* __restrict__ bias,
                            float* __restrict__ out,
                            int kn_shift, int w, int i_dim, int K_pad,
                            int start, int gbase)
{
    __shared__ __align__(16) __bf16 aTile[128 * LDS_K];  // [row][k]
    __shared__ __align__(16) __bf16 bTile[128 * LDS_K];  // [n][k] (transposed)

    const int knm1   = (1 << kn_shift) - 1;

    const int t      = threadIdx.x;
    const int lane   = t & 31;
    const int wid    = t >> 5;
    const int wave_m = wid & 3;       // 4 waves x 32 rows
    const int wave_n = wid >> 2;      // 2 waves x 64 cols

    const int blockM = blockIdx.x * 128;
    const int blockN = blockIdx.y * 128;

    // A staging: 256 threads -> 128 rows x 2 segments of 16 K
    const int  ar    = t >> 1;
    const int  aseg  = (t & 1) * 16;
    const int  Rrow  = blockM + ar;
    const int  ab    = Rrow >> kn_shift;        // batch   (kn is pow2)
    const int  ak    = Rrow & knm1;             // group
    const long abase = (long)ab * TOTAL_IN + start + (long)ak * w;
    const long abias = (long)ab * TOTAL_IN + start + ((long)w << kn_shift) + ak;

    // B staging: 256 threads -> 128 cols x 2 segments of 16 K
    const int bn   = t & 127;
    const int bkh  = (t >> 7) * 16;
    const int bcol = blockN + bn;

    // ---- accumulators initialized with bias[ocol]: C = A@W + bias in one pass ----
    const int ocl = lane & 15;
    v8f acc[2][4];
#pragma unroll
    for (int ni = 0; ni < 4; ++ni) {
        const int   ocol = blockN + wave_n * 64 + ni * 16 + ocl;
        const float bv   = bias[ocol];
#pragma unroll
        for (int mi = 0; mi < 2; ++mi)
#pragma unroll
            for (int j = 0; j < 8; ++j)
                acc[mi][ni][j] = bv;
    }

    const int nk = K_pad >> 5;

    // ---- software pipeline: registers hold chunk kt+1 while LDS holds kt ----
    float va[16], vb[16];
    load_a16(x, abase, abias, w, aseg, /*tail=*/(nk == 1), va);
    load_b16(W, i_dim, bcol,     bkh,  /*tail=*/(nk == 1), vb);

    for (int kt = 0; kt < nk; ++kt) {
        // ---- pack f32 -> bf16 and store 2x b128 per tile to LDS ----
        {
            union { __bf16 h[16]; uint4 q[2]; } ua, ub;
#pragma unroll
            for (int i = 0; i < 16; ++i) ua.h[i] = (__bf16)va[i];
#pragma unroll
            for (int i = 0; i < 16; ++i) ub.h[i] = (__bf16)vb[i];
            uint4* pa = (uint4*)&aTile[ar * LDS_K + aseg];
            uint4* pb = (uint4*)&bTile[bn * LDS_K + bkh];
            pa[0] = ua.q[0]; pa[1] = ua.q[1];
            pb[0] = ub.q[0]; pb[1] = ub.q[1];
        }
        __syncthreads();

        // ---- issue next chunk's global loads (overlap with ds_load + wmma) ----
        if (kt + 1 < nk) {
            load_a16(x, abase, abias, w, (kt + 1) * 32 + aseg, (kt + 2 == nk), va);
            load_b16(W, i_dim, bcol,     (kt + 1) * 32 + bkh,  (kt + 2 == nk), vb);
        }

        // ---- fragments from LDS (ISA VGPR layouts) ----
        v16bf afrag[2], bfrag[4];
        const int am    = lane & 15;
        const int akoff = (lane >> 4) * 8;
#pragma unroll
        for (int mi = 0; mi < 2; ++mi) {
            const __bf16* p = &aTile[(wave_m * 32 + mi * 16 + am) * LDS_K + akoff];
            const v8bf lo = *(const v8bf*)(p);        // K = koff..koff+7
            const v8bf hi = *(const v8bf*)(p + 16);   // K = koff+16..koff+23
            afrag[mi] = __builtin_shufflevector(lo, hi,
                0,1,2,3,4,5,6,7,8,9,10,11,12,13,14,15);
        }
        const int bnn   = lane & 15;
        const int bkoff = (lane >> 4) * 8;
#pragma unroll
        for (int ni = 0; ni < 4; ++ni) {
            const __bf16* p = &bTile[(wave_n * 64 + ni * 16 + bnn) * LDS_K + bkoff];
            const v8bf lo = *(const v8bf*)(p);
            const v8bf hi = *(const v8bf*)(p + 16);
            bfrag[ni] = __builtin_shufflevector(lo, hi,
                0,1,2,3,4,5,6,7,8,9,10,11,12,13,14,15);
        }

        // ---- 8 WMMAs: 32x64 wave tile, K += 32 ----
#pragma unroll
        for (int mi = 0; mi < 2; ++mi)
#pragma unroll
            for (int ni = 0; ni < 4; ++ni)
                acc[mi][ni] = __builtin_amdgcn_wmma_f32_16x16x32_bf16(
                    false, afrag[mi], false, bfrag[ni],
                    (short)0, acc[mi][ni], false, false);

        __syncthreads();
    }

    // ---- epilogue: scatter into concatenated (256, 240, 1024) output ----
    // Within an acc tile the 8 rows stay inside one batch (kn >= 16, 8-row aligned
    // window), so bb is constant and the address is base + v*DMODEL.
    const int rlo = (lane >> 4) * 8;
#pragma unroll
    for (int mi = 0; mi < 2; ++mi) {
        const int R0  = blockM + wave_m * 32 + mi * 16 + rlo;  // first row of this half
        const int bb  = R0 >> kn_shift;
        const int kk0 = R0 & knm1;
        const long rowbase = ((long)bb * NGROUPS + gbase + kk0) * DMODEL;
#pragma unroll
        for (int ni = 0; ni < 4; ++ni) {
            const int  ocol = blockN + wave_n * 64 + ni * 16 + ocl;
            float* __restrict__ po = out + rowbase + ocol;
#pragma unroll
            for (int v = 0; v < 8; ++v)
                po[(long)v * DMODEL] = acc[mi][ni][v];   // immediate-offset b32 stores
        }
    }
}

extern "C" void kernel_launch(void* const* d_in, const int* in_sizes, int n_in,
                              void* d_out, int out_size, void* d_ws, size_t ws_size,
                              hipStream_t stream) {
    (void)in_sizes; (void)n_in; (void)out_size; (void)d_ws; (void)ws_size;

    const float* x   = (const float*)d_in[0];
    float*       out = (float*)d_out;

    // Per-layer: w = ks*ci, i_dim = w+1, kn (pow2), start, gbase, K_pad (mult of 32)
    static const int W_[4]      = {27, 144, 288, 576};
    static const int IDIM_[4]   = {28, 145, 289, 577};
    static const int KNSH_[4]   = {4, 5, 6, 7};          // kn = 16,32,64,128
    static const int START_[4]  = {0, 448, 5088, 23584};
    static const int GBASE_[4]  = {0, 16, 48, 112};
    static const int KPAD_[4]   = {32, 160, 320, 608};

    for (int li = 0; li < 4; ++li) {
        const float* Wp = (const float*)d_in[1 + 3 * li];
        const float* bp = (const float*)d_in[2 + 3 * li];
        const int M = 256 << KNSH_[li];
        dim3 grid(M / 128, DMODEL / 128);
        dim3 block(256);
        fused_gather_gemm_bf16<<<grid, block, 0, stream>>>(
            x, Wp, bp, out,
            KNSH_[li], W_[li], IDIM_[li], KPAD_[li], START_[li], GBASE_[li]);
    }
}
